// NormFlowHomeomorphism_34626026340356
// MI455X (gfx1250) — matrix operand order repaired and checked by reading, not compiled, not run on hardware
//
#include <hip/hip_runtime.h>

// MI455X roofline: ~584 GFLOP of 64x64 GEMM vs ~34 MB HBM traffic
// (17 kFLOP/byte) -> purely compute-bound. Use v_wmma_f32_16x16x32_f16
// (f16 in / f32 accumulate). Weights (<1 MB) live in L2 (192 MB).
//
// All LDS regions are wave-private; CDNA5 LDS ops from one wave complete
// in order (ISA 7.3), so no s_barrier is needed -- only a compile-time
// fence to stop the compiler reordering DS accesses. This lets the 8
// waves per block slip phase and overlap WMMA/VALU/LDS across the WGP.

#define NLAYERS 32
#define HROW 24   // per-wave hbuf region is 64*HROW halves (3 KB); tile rows use stride 64

typedef __attribute__((ext_vector_type(16))) _Float16 v16h;
typedef __attribute__((ext_vector_type(8)))  float    v8f;

union AFrag { v16h v; float4 q[2]; _Float16 e[16]; };
union BFrag { v16h v; float4 q[2]; _Float16 e[16]; };
union Acc   { v8f  v; float f[8]; float4 q[2]; };

__device__ __forceinline__ void wave_fence() {
  // Compile-time ordering only: per-wave LDS ops are in-order in HW.
  asm volatile("" ::: "memory");
  __builtin_amdgcn_wave_barrier();
}

// Convert W2 (32x64x64) to f16 row-major, and W3 (32x2x64) to a zero-padded
// f16 32x16x64 so the 64->2 projection can ride the WMMA pipe too.
__global__ void prep_weights_f16(const float* __restrict__ W2,
                                 const float* __restrict__ W3,
                                 _Float16* __restrict__ w2h,
                                 _Float16* __restrict__ w3h) {
  int i = blockIdx.x * blockDim.x + threadIdx.x;
  if (i < NLAYERS * 64 * 64) w2h[i] = (_Float16)W2[i];
  if (i < NLAYERS * 16 * 64) {
    int L = i >> 10;          // / (16*64)
    int r = (i >> 6) & 15;    // padded row 0..15 (only 0,1 real)
    int k = i & 63;
    float v = (r < 2) ? W3[(L * 2 + r) * 64 + k] : 0.0f;
    w3h[i] = (_Float16)v;
  }
}

__global__ void __launch_bounds__(256)
flow_kernel(const float* __restrict__ x,
            const float* __restrict__ W1,
            const float* __restrict__ b1,
            const float* __restrict__ b2,
            const float* __restrict__ b3,
            const _Float16* __restrict__ w2h,
            const _Float16* __restrict__ w3h,
            float* __restrict__ out, int npts) {
  // Per-wave private LDS regions (8 waves per block).
  __shared__ __attribute__((aligned(16))) float z1buf[8 * 16];
  __shared__ __attribute__((aligned(16))) float z2buf[8 * 16];
  __shared__ __attribute__((aligned(32))) _Float16 hbuf[8 * 64 * HROW];

  const int lane  = threadIdx.x & 31;
  const int wave  = threadIdx.x >> 5;
  const int lm    = lane & 15;   // point index within tile / M row within A tile
  const int lhalf = lane >> 4;   // A-operand lane-half select
  const int wslot = wave * 16;
  const int woff  = wave * (64 * HROW);

  int p0 = blockIdx.x * 128 + wslot + lm;
  int p  = (p0 < npts) ? p0 : (npts - 1);   // clamp: keep EXEC full for WMMA

  if (lane < 16) {
    float2 z = reinterpret_cast<const float2*>(x)[p];
    z1buf[wslot + lm] = z.x;
    z2buf[wslot + lm] = z.y;
  }
  wave_fence();

  for (int L = 0; L < NLAYERS; ++L) {
    // ---- broadcast z1 of all 16 points to every lane (b128 LDS reads)
    float z1s[16];
#pragma unroll
    for (int i = 0; i < 4; ++i) {
      float4 q = reinterpret_cast<const float4*>(z1buf + wslot)[i];
      z1s[4 * i + 0] = q.x; z1s[4 * i + 1] = q.y;
      z1s[4 * i + 2] = q.z; z1s[4 * i + 3] = q.w;
    }

    // ---- layer 1: h = relu(z1 * W1^T + b1); lane owns point n = lm, elements
    //      are K (table order: 32c + 8*lhalf + s, +8 for s >= 8).
    BFrag B[2];
#pragma unroll
    for (int c = 0; c < 2; ++c) {
#pragma unroll
      for (int s = 0; s < 16; ++s) {
        int k = L * 64 + 32 * c + 8 * lhalf + (s < 8 ? s : s + 8);
        B[c].e[s] = (_Float16)fmaxf(fmaf(z1s[lm], W1[k], b1[k]), 0.0f);
      }
    }

    // ---- layer 2: H2 = relu(W2 @ H1 + b2) : 4 M-tiles x 2 K-chunks = 8 WMMAs.
    //      b2 folded into C (D layout: reg r of lane -> m = 16t + 8*lhalf + r).
    const float* b2L = b2 + L * 64;
    Acc acc[4];
#pragma unroll
    for (int t = 0; t < 4; ++t) {
      int m0 = 16 * t + 8 * lhalf;
      acc[t].q[0] = reinterpret_cast<const float4*>(b2L + m0)[0];
      acc[t].q[1] = reinterpret_cast<const float4*>(b2L + m0)[1];
    }
#pragma unroll
    for (int c = 0; c < 2; ++c) {
#pragma unroll
      for (int t = 0; t < 4; ++t) {
        // A layout: lane half selects K 0-7/16-23 vs 8-15/24-31 within chunk.
        const _Float16* row = w2h + ((L * 64 + 16 * t + lm) * 64) + (32 * c + 8 * lhalf);
        AFrag a;
        a.q[0] = reinterpret_cast<const float4*>(row)[0];  // K base..base+7
        a.q[1] = reinterpret_cast<const float4*>(row)[2];  // K base+16..base+23
        acc[t].v = __builtin_amdgcn_wmma_f32_16x16x32_f16(
            false, a.v, false, B[c].v, (short)0, acc[t].v, false, false);
      }
    }

    // ---- relu, cvt to f16, and transpose D(lane=n,reg=m) -> B(lane=n,reg=k)
    //      via per-wave LDS tile (point-major rows, stride 64).
#pragma unroll
    for (int t = 0; t < 4; ++t) {
      int mb = 16 * t + 8 * lhalf;
#pragma unroll
      for (int r = 0; r < 8; ++r)
        hbuf[woff + lm * 64 + mb + r] = (_Float16)fmaxf(acc[t].f[r], 0.0f);
    }
    wave_fence();
    BFrag B2[2];
#pragma unroll
    for (int c = 0; c < 2; ++c) {  // lane serves point lm, K table order in chunk c
      const float4* rp =
          reinterpret_cast<const float4*>(&hbuf[woff + lm * 64 + 32 * c + 8 * lhalf]);
      B2[c].q[0] = rp[0];
      B2[c].q[1] = rp[2];
    }

    // ---- layer 3: out = W3pad @ H2 : 2 WMMAs; shift/log_scale land in
    //      regs 0/1 of lanes 0..15 (D: m=0,1 rows).
    Acc o;
#pragma unroll
    for (int r = 0; r < 8; ++r) o.f[r] = 0.0f;
#pragma unroll
    for (int c = 0; c < 2; ++c) {
      const _Float16* row3 = w3h + ((L * 16 + lm) * 64) + (32 * c + 8 * lhalf);
      AFrag a3;
      a3.q[0] = reinterpret_cast<const float4*>(row3)[0];
      a3.q[1] = reinterpret_cast<const float4*>(row3)[2];
      o.v = __builtin_amdgcn_wmma_f32_16x16x32_f16(
          false, a3.v, false, B2[c].v, (short)0, o.v, false, false);
    }

    // ---- coupling: z2n = z2*exp(log_scale)+shift; swap halves
    if (lane < 16) {
      float shift = o.f[0] + b3[L * 2 + 0];
      float lsc   = o.f[1] + b3[L * 2 + 1];
      float z2v   = z2buf[wslot + lm];
      float z1o   = z1s[lm];
      float z2n   = fmaf(z2v, __expf(lsc), shift);
      z1buf[wslot + lm] = z2n;   // new z1
      z2buf[wslot + lm] = z1o;   // new z2
    }
    wave_fence();
  }

  if (lane < 16 && p0 < npts) {
    float2 o2;
    o2.x = z1buf[wslot + lm];
    o2.y = z2buf[wslot + lm];
    reinterpret_cast<float2*>(out)[p] = o2;
  }
}

extern "C" void kernel_launch(void* const* d_in, const int* in_sizes, int n_in,
                              void* d_out, int out_size, void* d_ws, size_t ws_size,
                              hipStream_t stream) {
  const float* x  = (const float*)d_in[0];
  const float* W1 = (const float*)d_in[1];
  const float* b1 = (const float*)d_in[2];
  const float* W2 = (const float*)d_in[3];
  const float* b2 = (const float*)d_in[4];
  const float* W3 = (const float*)d_in[5];
  const float* b3 = (const float*)d_in[6];

  _Float16* w2h = (_Float16*)d_ws;                       // 32*64*64 f16 = 256 KB
  _Float16* w3h = w2h + NLAYERS * 64 * 64;               // 32*16*64 f16 =  64 KB

  int npts = in_sizes[0] / 2;

  int prep_total = NLAYERS * 64 * 64;
  prep_weights_f16<<<(prep_total + 255) / 256, 256, 0, stream>>>(W2, W3, w2h, w3h);

  int nblocks = (npts + 127) / 128;
  flow_kernel<<<nblocks, 256, 0, stream>>>(x, W1, b1, b2, b3, w2h, w3h,
                                           (float*)d_out, npts);
}
